// MultiHeadAttn_12773232739003
// MI455X (gfx1250) — matrix operand order repaired and checked by reading, not compiled, not run on hardware
//
#include <hip/hip_runtime.h>
#include <hip/hip_bf16.h>

typedef __attribute__((ext_vector_type(16))) __bf16 v16bf;
typedef __attribute__((ext_vector_type(8)))  float  v8f;
typedef __attribute__((ext_vector_type(4)))  unsigned int u32x4;
typedef __attribute__((ext_vector_type(8)))  int i32x8;
typedef __attribute__((ext_vector_type(4)))  int i32x4;

#if defined(__has_builtin)
#  if __has_builtin(__builtin_amdgcn_tensor_load_to_lds) && \
      __has_builtin(__builtin_amdgcn_s_wait_tensorcnt)
#    define USE_TDM 1
#  endif
#endif
#ifndef USE_TDM
#  define USE_TDM 0
#endif

#define S_LEN 2048
#define HIDN  2048
#define NHEAD 16
#define DHEAD 128

#define BM 128
#define BN 64
#define BK 64
#define LDT 72      // padded LDS row stride (bf16): 144B = 36 banks -> conflict-free frags
#define KLD 136     // K-tile LDS stride (d=128 + 8 pad)
#define VLD 72      // Vt-tile LDS stride (kpos=64 + 8 pad)
#define PLD 72      // P-tile LDS stride

union F16u { uint4 u[2]; v16bf v; };

// A-operand fragment (16x32 bf16): lane half K-interleave {0..7,16..23} / {8..15,24..31}
__device__ __forceinline__ v16bf load_fragA(const __bf16* row, int half8) {
  F16u f;
  f.u[0] = *(const uint4*)(row + half8);
  f.u[1] = *(const uint4*)(row + half8 + 16);
  return f.v;
}
// B-operand fragment (32x16 bf16): lane half K-interleave {0..7,16..23} / {8..15,24..31}
__device__ __forceinline__ v16bf load_fragB(const __bf16* row, int k16) {
  F16u f;
  f.u[0] = *(const uint4*)(row + (k16 >> 1));
  f.u[1] = *(const uint4*)(row + (k16 >> 1) + 16);
  return f.v;
}
__device__ __forceinline__ v8f wmma_bf16(v16bf a, v16bf b, v8f c) {
  return __builtin_amdgcn_wmma_f32_16x16x32_bf16(false, a, false, b, (short)0, c,
                                                 false, false);
}
__device__ __forceinline__ unsigned short bfbits(float f) {
  union { __bf16 b; unsigned short u; } cv; cv.b = (__bf16)f; return cv.u;
}

#if USE_TDM
// Issue a 2-D bf16 tile DMA: rows x cols elements from row-major global (row
// stride = srcStride elems) into LDS at ldsOff, with LDS padding of padDwords
// every (1<<ivalCode)*2 DWORDs.  Descriptor per CDNA5 ISA section 8.3/8.4.
__device__ __forceinline__ void tdm_load_tile_bf16(const __bf16* src, unsigned ldsOff,
                                                   int rows, int cols, int srcStride,
                                                   int ivalCode, int padDwords,
                                                   int tensorRows) {
  const unsigned long long ga = (unsigned long long)(const void*)src;
  u32x4 g0;
  g0[0] = 1u;                                   // count=1, user descriptor
  g0[1] = ldsOff;                               // lds_addr (bytes)
  g0[2] = (unsigned)(ga & 0xffffffffu);         // global_addr[31:0]
  g0[3] = (unsigned)((ga >> 32) & 0x01ffffffu)  // global_addr[56:32]
        | 0x80000000u;                          // type = 2 ("image")
  i32x8 g1;
  g1[0] = (1 << 16)                              // data_size = 2 bytes
        | (1 << 20)                              // pad_enable
        | (ivalCode << 22)                       // pad_interval code
        | ((padDwords - 1) << 25);               // pad_amount code (0 => 1 DWORD)
  g1[1] = (cols & 0xffff) << 16;                 // tensor_dim0[15:0]
  g1[2] = (tensorRows & 0xffff) << 16;           // tensor_dim1[15:0] (dim0 hi = 0)
  g1[3] = (cols & 0xffff) << 16;                 // tile_dim0 (dim1 hi = 0)
  g1[4] = rows & 0xffff;                         // tile_dim1 (tile_dim2 = 0)
  g1[5] = srcStride;                             // tensor_dim0_stride[31:0]
  g1[6] = 0;                                     // dim0_stride hi, dim1_stride lo
  g1[7] = 0;                                     // dim1_stride hi
  const i32x4 gz4 = {0, 0, 0, 0};
  const i32x8 gz8 = {0, 0, 0, 0, 0, 0, 0, 0};
  __builtin_amdgcn_tensor_load_to_lds(g0, g1, gz4, gz4, gz8, 0);
}
#endif

// ---------------- generic GEMM core: C[128x64] += A[M,2048] @ W[N,2048]^T ----------------
template<bool AF32>
__device__ __forceinline__ void gemm_tile(const void* Aptr, const float* W,
                                          int mBase, int nBase, v8f acc[2][2]) {
  __shared__ __bf16 Alds[BM * LDT];
  __shared__ __bf16 Wlds[BN * LDT];
  const int tid   = threadIdx.x;
  const int lane  = tid & 31;
  const int wave  = tid >> 5;
  const int ln    = lane & 15;
  const int half8 = (lane >> 4) << 3;
  const int k16   = (lane >> 4) << 4;
  const int wm    = wave >> 1;   // 0..3 -> 32-row strip
  const int wn    = wave & 1;    // 0..1 -> 32-col strip
#if USE_TDM
  const int wv = __builtin_amdgcn_readfirstlane(tid) >> 5;  // scalar wave id
#endif

  for (int kb = 0; kb < HIDN; kb += BK) {
    if (AF32) {
      const float* A = (const float*)Aptr;
      #pragma unroll
      for (int it = 0; it < 8; ++it) {
        int idx = tid + it * 256;
        int r = idx >> 4, c4 = idx & 15;
        float4 f = *(const float4*)(A + (size_t)(mBase + r) * HIDN + kb + c4 * 4);
        uint2 u;
        u.x = (unsigned)bfbits(f.x) | ((unsigned)bfbits(f.y) << 16);
        u.y = (unsigned)bfbits(f.z) | ((unsigned)bfbits(f.w) << 16);
        *(uint2*)&Alds[r * LDT + c4 * 4] = u;
      }
    } else {
#if USE_TDM
      // 128x64 bf16 tile via Tensor Data Mover; LDS pad 4 DWORDs per 32 DWORDs
      // reproduces the padded stride LDT=72.  Wave 0 issues; EXEC is ignored by
      // TDM so a scalar branch (readfirstlane) avoids duplicate DMAs.
      if (wv == 0) {
        const __bf16* A = (const __bf16*)Aptr;
        tdm_load_tile_bf16(A + (size_t)mBase * HIDN + kb,
                           (unsigned)(size_t)(const void*)Alds,
                           BM, BK, HIDN, /*ival 32 DWORDs*/ 4, /*pad*/ 4, 4096);
      }
#else
      const __bf16* A = (const __bf16*)Aptr;
      #pragma unroll
      for (int it = 0; it < 4; ++it) {
        int idx = tid + it * 256;
        int r = idx >> 3, c8 = idx & 7;
        *(uint4*)&Alds[r * LDT + c8 * 8] =
            *(const uint4*)(A + (size_t)(mBase + r) * HIDN + kb + c8 * 8);
      }
#endif
    }
    #pragma unroll
    for (int it = 0; it < 4; ++it) {
      int idx = tid + it * 256;
      int r = idx >> 4, c4 = idx & 15;
      float4 f = *(const float4*)(W + (size_t)(nBase + r) * HIDN + kb + c4 * 4);
      uint2 u;
      u.x = (unsigned)bfbits(f.x) | ((unsigned)bfbits(f.y) << 16);
      u.y = (unsigned)bfbits(f.z) | ((unsigned)bfbits(f.w) << 16);
      *(uint2*)&Wlds[r * LDT + c4 * 4] = u;
    }
#if USE_TDM
    if (!AF32 && wv == 0) __builtin_amdgcn_s_wait_tensorcnt(0);
#endif
    __syncthreads();

    if (kb + BK < HIDN) {   // global_prefetch_b8 of next K-chunk
      if (AF32) {
        const float* A = (const float*)Aptr;
        __builtin_prefetch(A + (size_t)(mBase + (tid >> 1)) * HIDN + kb + BK + (tid & 1) * 32, 0, 1);
      }
      if (tid < 128)
        __builtin_prefetch(W + (size_t)(nBase + (tid >> 1)) * HIDN + kb + BK + (tid & 1) * 32, 0, 1);
    }

    #pragma unroll
    for (int kk = 0; kk < BK; kk += 32) {
      v16bf af[2], bfr[2];
      #pragma unroll
      for (int i = 0; i < 2; ++i)
        af[i] = load_fragA(&Alds[(wm * 32 + i * 16 + ln) * LDT + kk], half8);
      #pragma unroll
      for (int j = 0; j < 2; ++j)
        bfr[j] = load_fragB(&Wlds[(wn * 32 + j * 16 + ln) * LDT + kk], k16);
      #pragma unroll
      for (int i = 0; i < 2; ++i)
        #pragma unroll
        for (int j = 0; j < 2; ++j)
          acc[i][j] = wmma_bf16(af[i], bfr[j], acc[i][j]);
    }
    __syncthreads();
  }
}

// ---------------- kernel 1: fused QKV projection, bf16 out in [B,H,S,D] ----------------
__global__ __launch_bounds__(256) void qkv_kernel(
    const float* __restrict__ X,
    const float* __restrict__ Wq, const float* __restrict__ bq,
    const float* __restrict__ Wk, const float* __restrict__ bk,
    const float* __restrict__ Wv, const float* __restrict__ bv,
    __bf16* __restrict__ Qb, __bf16* __restrict__ Kb, __bf16* __restrict__ Vb) {
  const int mat = blockIdx.y >> 5;
  const int nb  = blockIdx.y & 31;
  const float* W    = (mat == 0) ? Wq : (mat == 1) ? Wk : Wv;
  const float* bias = (mat == 0) ? bq : (mat == 1) ? bk : bv;
  __bf16*      dst  = (mat == 0) ? Qb : (mat == 1) ? Kb : Vb;
  const int mBase = blockIdx.x * BM;
  const int nBase = nb * BN;

  v8f acc[2][2] = {};
  gemm_tile<true>(X, W, mBase, nBase, acc);

  const int tid = threadIdx.x, lane = tid & 31, wave = tid >> 5;
  const int ln = lane & 15, mh = (lane >> 4) << 3;
  const int wm = wave >> 1, wn = wave & 1;
  #pragma unroll
  for (int j = 0; j < 2; ++j) {
    const int col = nBase + wn * 32 + j * 16 + ln;
    const float bias_v = bias[col];
    const int hh = col >> 7, d = col & 127;
    #pragma unroll
    for (int i = 0; i < 2; ++i) {
      #pragma unroll
      for (int r = 0; r < 8; ++r) {
        const int m = mBase + wm * 32 + i * 16 + r + mh;
        const int bidx = m >> 11, s = m & 2047;
        dst[(((size_t)(bidx * NHEAD + hh)) * S_LEN + s) * DHEAD + d] =
            (__bf16)(acc[i][j][r] + bias_v);
      }
    }
  }
}

// ---------------- kernel 2: causal flash attention, bf16 H out in [B,S,HID] ----------------
__global__ __launch_bounds__(128) void attn_kernel(
    const __bf16* __restrict__ Q, const __bf16* __restrict__ K,
    const __bf16* __restrict__ V, __bf16* __restrict__ H) {
  __shared__ __bf16 Klds[64 * KLD];     // [kpos][d]
  __shared__ __bf16 Vtlds[128 * VLD];   // [d][kpos] (transposed during staging)
  __shared__ __bf16 Plds[4 * 16 * PLD]; // wave-private P tile (C-layout -> A-layout relay)
  const int tid = threadIdx.x, lane = tid & 31, wave = tid >> 5;
  const int ln = lane & 15;
  const int half8 = (lane >> 4) << 3;
  const int k16 = (lane >> 4) << 4;
  const int qb = blockIdx.x, bh = blockIdx.y;
  const int b = bh >> 4, h = bh & 15;
  const size_t hoff = (size_t)bh * S_LEN * DHEAD;
  const int q0 = qb * 64 + wave * 16;
  const float NEGINF = -__builtin_inff();
  const float scale = 0.08838834764831845f;   // 1/sqrt(128)
#if USE_TDM
  const int wv = __builtin_amdgcn_readfirstlane(tid) >> 5;
#endif

  v16bf qf[4];
  {
    const __bf16* qrow = Q + hoff + (size_t)(q0 + ln) * DHEAD;
    #pragma unroll
    for (int ds = 0; ds < 4; ++ds) qf[ds] = load_fragA(qrow + ds * 32, half8);
  }
  v8f oacc[8] = {};
  float m_r[8], l_r[8];
  #pragma unroll
  for (int j = 0; j < 8; ++j) { m_r[j] = NEGINF; l_r[j] = 0.f; }

  __bf16* Pw = &Plds[wave * 16 * PLD];

  for (int kb = 0; kb <= qb; ++kb) {
    const __bf16* Kg = K + hoff + (size_t)kb * 64 * DHEAD;
    const __bf16* Vg = V + hoff + (size_t)kb * 64 * DHEAD;
#if USE_TDM
    // K tile 64x128 bf16 via TDM; LDS pad 4 DWORDs per 64 DWORDs -> stride KLD=136.
    if (wv == 0)
      tdm_load_tile_bf16(Kg, (unsigned)(size_t)(const void*)Klds,
                         64, DHEAD, DHEAD, /*ival 64 DWORDs*/ 5, /*pad*/ 4, S_LEN);
#else
    #pragma unroll
    for (int it = 0; it < 8; ++it) {           // stage K tile 64x128
      int idx = tid + it * 128;
      int r = idx >> 4, c = idx & 15;
      *(uint4*)&Klds[r * KLD + c * 8] = *(const uint4*)(Kg + r * DHEAD + c * 8);
    }
#endif
    #pragma unroll
    for (int it = 0; it < 32; ++it) {          // stage V tile transposed -> [d][kpos]
      int idx = tid + it * 128;
      int r = idx >> 6, cu = idx & 63;
      unsigned int u = *(const unsigned int*)(Vg + r * DHEAD + cu * 2);
      union { unsigned short s; __bf16 b; } lo, hi;
      lo.s = (unsigned short)(u & 0xffffu);
      hi.s = (unsigned short)(u >> 16);
      Vtlds[(cu * 2 + 0) * VLD + r] = lo.b;
      Vtlds[(cu * 2 + 1) * VLD + r] = hi.b;
    }
#if USE_TDM
    if (wv == 0) __builtin_amdgcn_s_wait_tensorcnt(0);
#endif
    __syncthreads();
    if (kb < qb) {
      __builtin_prefetch(Kg + (size_t)64 * DHEAD + (size_t)tid * 64, 0, 1);
      __builtin_prefetch(Vg + (size_t)64 * DHEAD + (size_t)tid * 64, 0, 1);
    }

    v8f sc[4] = {};                            // scores: 16q x 64k
    #pragma unroll
    for (int t = 0; t < 4; ++t)
      #pragma unroll
      for (int ds = 0; ds < 4; ++ds) {
        v16bf kf = load_fragB(&Klds[(t * 16 + ln) * KLD + ds * 32], k16);
        sc[t] = wmma_bf16(qf[ds], kf, sc[t]);
      }

    const bool diag = (kb == qb);
    float alpha[8];
    #pragma unroll
    for (int j = 0; j < 8; ++j) {              // online softmax, fp32
      const int qg = q0 + j + half8;
      float mt = NEGINF;
      #pragma unroll
      for (int t = 0; t < 4; ++t) {
        float v = sc[t][j] * scale;
        if (diag) {
          const int kg = kb * 64 + t * 16 + ln;
          if (kg > qg) v = NEGINF;
        }
        sc[t][j] = v;
        mt = fmaxf(mt, v);
      }
      #pragma unroll
      for (int x = 1; x < 16; x <<= 1) mt = fmaxf(mt, __shfl_xor(mt, x, 32));
      const float mn = fmaxf(m_r[j], mt);
      alpha[j] = __expf(m_r[j] - mn);
      m_r[j] = mn;
      float rs = 0.f;
      #pragma unroll
      for (int t = 0; t < 4; ++t) {
        const float p = __expf(sc[t][j] - mn);
        sc[t][j] = p;
        rs += p;
      }
      #pragma unroll
      for (int x = 1; x < 16; x <<= 1) rs += __shfl_xor(rs, x, 32);
      l_r[j] = l_r[j] * alpha[j] + rs;
    }
    #pragma unroll
    for (int dt = 0; dt < 8; ++dt)
      #pragma unroll
      for (int j = 0; j < 8; ++j) oacc[dt][j] *= alpha[j];

    #pragma unroll
    for (int j = 0; j < 8; ++j)                // P: C-layout -> LDS (row-major bf16)
      #pragma unroll
      for (int t = 0; t < 4; ++t)
        Pw[(j + half8) * PLD + t * 16 + ln] = (__bf16)sc[t][j];

    #pragma unroll
    for (int kk = 0; kk < 64; kk += 32) {      // O += P @ V
      v16bf pf = load_fragA(&Pw[ln * PLD + kk], half8);
      #pragma unroll
      for (int dt = 0; dt < 8; ++dt) {
        v16bf vf = load_fragB(&Vtlds[(dt * 16 + ln) * VLD + kk], k16);
        oacc[dt] = wmma_bf16(pf, vf, oacc[dt]);
      }
    }
    __syncthreads();
  }

  float inv_l[8];
  #pragma unroll
  for (int j = 0; j < 8; ++j) inv_l[j] = 1.f / l_r[j];
  #pragma unroll
  for (int dt = 0; dt < 8; ++dt) {
    const int d = dt * 16 + ln;
    #pragma unroll
    for (int j = 0; j < 8; ++j) {
      const int q = q0 + j + half8;
      H[((size_t)b * S_LEN + q) * HIDN + h * DHEAD + d] = (__bf16)(oacc[dt][j] * inv_l[j]);
    }
  }
}

// ---------------- kernel 3: output projection, fp32 out ----------------
__global__ __launch_bounds__(256) void oproj_kernel(
    const __bf16* __restrict__ H, const float* __restrict__ Wo,
    const float* __restrict__ bo, float* __restrict__ out) {
  const int mBase = blockIdx.x * BM;
  const int nBase = blockIdx.y * BN;
  v8f acc[2][2] = {};
  gemm_tile<false>(H, Wo, mBase, nBase, acc);

  const int tid = threadIdx.x, lane = tid & 31, wave = tid >> 5;
  const int ln = lane & 15, mh = (lane >> 4) << 3;
  const int wm = wave >> 1, wn = wave & 1;
  #pragma unroll
  for (int j = 0; j < 2; ++j) {
    const int col = nBase + wn * 32 + j * 16 + ln;
    const float bias_v = bo[col];
    #pragma unroll
    for (int i = 0; i < 2; ++i)
      #pragma unroll
      for (int r = 0; r < 8; ++r) {
        const int m = mBase + wm * 32 + i * 16 + r + mh;
        out[(size_t)m * HIDN + col] = acc[i][j][r] + bias_v;
      }
  }
}

extern "C" void kernel_launch(void* const* d_in, const int* in_sizes, int n_in,
                              void* d_out, int out_size, void* d_ws, size_t ws_size,
                              hipStream_t stream) {
  (void)in_sizes; (void)n_in; (void)out_size; (void)ws_size;
  const float* X  = (const float*)d_in[0];
  // d_in[1] = mask (all-ones; tril(ones) == causal, applied analytically in-kernel)
  const float* Wq = (const float*)d_in[2];
  const float* bq = (const float*)d_in[3];
  const float* Wk = (const float*)d_in[4];
  const float* bk = (const float*)d_in[5];
  const float* Wv = (const float*)d_in[6];
  const float* bv = (const float*)d_in[7];
  const float* Wo = (const float*)d_in[8];
  const float* bo = (const float*)d_in[9];
  float* out = (float*)d_out;

  const size_t MATE = (size_t)4096 * 2048;   // elements per bf16 intermediate
  __bf16* Qb = (__bf16*)d_ws;
  __bf16* Kb = Qb + MATE;
  __bf16* Vb = Kb + MATE;
  __bf16* Hb = Vb + MATE;

  qkv_kernel<<<dim3(32, 96), 256, 0, stream>>>(X, Wq, bq, Wk, bk, Wv, bv, Qb, Kb, Vb);
  attn_kernel<<<dim3(32, 32), 128, 0, stream>>>(Qb, Kb, Vb, Hb);
  oproj_kernel<<<dim3(32, 32), 256, 0, stream>>>(Hb, Wo, bo, out);
}
